// MultiHeadAttention_13374528160133
// MI455X (gfx1250) — matrix operand order repaired and checked by reading, not compiled, not run on hardware
//
#include <hip/hip_runtime.h>

// ---------------------------------------------------------------------------
// MHA forward for MI455X (gfx1250, wave32, WMMA + TDM + async global->LDS).
//   proj:   q/k/v = X @ W_h           (f16 frags, f32 acc)  -> [B,H,S,DK] f16
//   attn:   flash-attention over S    (WMMA QK^T + WMMA PV) -> monolith f16
//   out:    monolith @ WO             (f32 out)
// ---------------------------------------------------------------------------

typedef __attribute__((ext_vector_type(16))) _Float16 v16h;
typedef __attribute__((ext_vector_type(4)))  _Float16 v4h;
typedef __attribute__((ext_vector_type(8)))  float    v8f;
typedef __attribute__((ext_vector_type(4)))  unsigned int v4u;
typedef __attribute__((ext_vector_type(8)))  int      v8i;
typedef __attribute__((ext_vector_type(4)))  int      v4i;

constexpr int kD  = 1024;    // d_model
constexpr int kH  = 16;      // heads
constexpr int kDK = 64;      // head dim
constexpr int kB  = 4;       // batch
constexpr int kS  = 2048;    // seq len
constexpr int kM  = kB * kS; // 8192 rows

// LDS strides (f16 elements); 72 f16 = 144 B rows -> 16 B aligned, padded
constexpr int kLdQ = 72;
constexpr int kLdS = 68;     // fp32 score tile stride

// ---- gfx1250 async global->LDS DMA (ASYNCcnt-tracked) ---------------------
__device__ __forceinline__ void async_copy_b128(unsigned lds_off, const void* gaddr) {
  asm volatile("global_load_async_to_lds_b128 %0, %1, off"
               :: "v"(lds_off), "v"(gaddr) : "memory");
}
__device__ __forceinline__ void wait_async() {
  asm volatile("s_wait_asynccnt 0x0" ::: "memory");
}
__device__ __forceinline__ unsigned lds_off(const void* p) {
  // shared-aperture flat address: low 32 bits are the LDS byte offset
  return (unsigned)(unsigned long long)p;
}

// ---- gfx1250 Tensor Data Mover: 2D f16 tile -> padded LDS -----------------
// tile_d0 f16 per row (row = tile_d0*2 bytes), tile_d1 rows, row stride in
// f16 elems. LDS rows padded +16B per 128B stored (pad_interval=4 -> 32 dw,
// pad_amount=3 -> 4 dw), matching the kLdQ=72 (144 B) LDS stride.
__device__ __forceinline__ void tdm_load_2d(unsigned lds_byte_off,
                                            const void* gaddr,
                                            unsigned tile_d0,
                                            unsigned tile_d1,
                                            unsigned row_stride_elems,
                                            unsigned tensor_d1) {
  const unsigned long long ga = (unsigned long long)gaddr;
  v4u g0;
  g0[0] = 1u;                                      // count=1 descriptor
  g0[1] = lds_byte_off;                            // lds_addr
  g0[2] = (unsigned)ga;                            // global_addr[31:0]
  g0[3] = (unsigned)(ga >> 32) | (2u << 30);       // global_addr[56:32], type=2
  const unsigned td0 = tile_d0;                    // tensor_dim0 == tile_dim0
  v8i g1;
  g1[0] = (int)((1u << 16)                         // data_size = 2 bytes
              | (1u << 20)                         // pad_enable
              | (4u << 22)                         // pad_interval: 32 dwords
              | (3u << 25));                       // pad_amount:   4 dwords
  g1[1] = (int)((td0 & 0xFFFFu) << 16);            // tensor_dim0[15:0]
  g1[2] = (int)((td0 >> 16) | ((tensor_d1 & 0xFFFFu) << 16));
  g1[3] = (int)((tensor_d1 >> 16) | (tile_d0 << 16)); // tile_dim0
  g1[4] = (int)(tile_d1 & 0xFFFFu);                // tile_dim1 (tile_dim2=0)
  g1[5] = (int)row_stride_elems;                   // tensor_dim0_stride[31:0]
  g1[6] = 0;
  g1[7] = 0;
  const v4i z4 = (v4i)0;
  const v8i z8 = (v8i)0;
  __builtin_amdgcn_tensor_load_to_lds(g0, g1, z4, z4, z8, 0);
}
__device__ __forceinline__ void wait_tensor() {
  __builtin_amdgcn_s_wait_tensorcnt(0);
}

// ---- WMMA fragment gathers (contiguous k = 16*(lane/16) + i in A and B, wave32) -------------------
__device__ __forceinline__ v16h frag_a(const _Float16* lds, int ld,
                                       int m_base, int k_base, int lane) {
  const int m  = m_base + (lane & 15);
  const int kb = (lane & 16) ? 16 : 0;
  v16h a;
#pragma unroll
  for (int i = 0; i < 16; ++i) {
    const int k = k_base + kb + i;
    a[i] = lds[m * ld + k];
  }
  return a;
}

// B 32x16 from n-major tile lds[n][k]
__device__ __forceinline__ v16h frag_b_nmajor(const _Float16* lds, int ld,
                                              int n_base, int k_base, int lane) {
  const int n  = n_base + (lane & 15);
  const int kb = k_base + ((lane & 16) ? 16 : 0);
  v16h b;
#pragma unroll
  for (int i = 0; i < 16; ++i) b[i] = lds[n * ld + kb + i];
  return b;
}

// B 32x16 from k-major tile lds[k][n]
__device__ __forceinline__ v16h frag_b_kmajor(const _Float16* lds, int ld,
                                              int n_base, int k_base, int lane) {
  const int n  = n_base + (lane & 15);
  const int kb = k_base + ((lane & 16) ? 16 : 0);
  v16h b;
#pragma unroll
  for (int i = 0; i < 16; ++i) b[i] = lds[(kb + i) * ld + n];
  return b;
}

// ---------------------------------------------------------------------------
// Kernel 1: per-head projections. grid = (kM/64, kH, 3), block = 128 (4 waves)
// 64x64 block tile, K staged 64 wide. fp32 tiles staged via float4 clauses.
// ---------------------------------------------------------------------------
__global__ void __launch_bounds__(128)
proj_kernel(const float* __restrict__ Q, const float* __restrict__ K,
            const float* __restrict__ V,
            const float* __restrict__ WQ, const float* __restrict__ WK,
            const float* __restrict__ WV,
            _Float16* __restrict__ qh, _Float16* __restrict__ kh,
            _Float16* __restrict__ vh) {
  __shared__ alignas(16) _Float16 ldsA[64 * kLdQ];
  __shared__ alignas(16) _Float16 ldsBt[64 * kLdQ];

  const float* X; const float* W; _Float16* Y;
  if (blockIdx.z == 0)      { X = Q; W = WQ; Y = qh; }
  else if (blockIdx.z == 1) { X = K; W = WK; Y = kh; }
  else                      { X = V; W = WV; Y = vh; }

  const int t    = threadIdx.x;
  const int wave = t >> 5, lane = t & 31;
  const int m0   = blockIdx.x * 64;
  const int h    = blockIdx.y;                 // 64-wide N tile == one head
  const float* Wh = W + (size_t)h * kD * kDK;  // [kD][kDK] row-major

  v8f acc[4] = {};

  for (int kk = 0; kk < kD; kk += 64) {
    // A tile 64x64 fp32: batched float4 loads, then cvt + packed b64 stores
    const float4* Xv = (const float4*)(X + (size_t)m0 * kD + kk);
    float4 ta[8];
#pragma unroll
    for (int i = 0; i < 8; ++i) {
      const int idx = t + i * 128;             // 1024 float4
      const int r = idx >> 4, c4 = idx & 15;
      ta[i] = Xv[(size_t)r * (kD / 4) + c4];
    }
#pragma unroll
    for (int i = 0; i < 8; ++i) {
      const int idx = t + i * 128;
      const int r = idx >> 4, c4 = idx & 15;
      v4h hv = { (_Float16)ta[i].x, (_Float16)ta[i].y,
                 (_Float16)ta[i].z, (_Float16)ta[i].w };
      *(v4h*)&ldsA[r * kLdQ + c4 * 4] = hv;
    }
    // W tile 64(d) x 64(n) fp32, stored n-major (transposed scatter)
    const float4* Wv = (const float4*)(Wh + (size_t)kk * kDK);
    float4 tb[8];
#pragma unroll
    for (int i = 0; i < 8; ++i) {
      const int idx = t + i * 128;
      const int d = idx >> 4, c4 = idx & 15;
      tb[i] = Wv[(size_t)d * (kDK / 4) + c4];
    }
#pragma unroll
    for (int i = 0; i < 8; ++i) {
      const int idx = t + i * 128;
      const int d = idx >> 4, c4 = idx & 15;
      ldsBt[(c4 * 4 + 0) * kLdQ + d] = (_Float16)tb[i].x;
      ldsBt[(c4 * 4 + 1) * kLdQ + d] = (_Float16)tb[i].y;
      ldsBt[(c4 * 4 + 2) * kLdQ + d] = (_Float16)tb[i].z;
      ldsBt[(c4 * 4 + 3) * kLdQ + d] = (_Float16)tb[i].w;
    }
    __syncthreads();

#pragma unroll
    for (int ks = 0; ks < 2; ++ks) {
      const v16h a = frag_a(ldsA, kLdQ, wave * 16, ks * 32, lane);
#pragma unroll
      for (int nt = 0; nt < 4; ++nt) {
        const v16h bfr = frag_b_nmajor(ldsBt, kLdQ, nt * 16, ks * 32, lane);
        acc[nt] = __builtin_amdgcn_wmma_f32_16x16x32_f16(
            false, a, false, bfr, (short)0, acc[nt], false, false);
      }
    }
    __syncthreads();
  }

  // epilogue -> [B,H,S,DK] f16
  const int mh = (lane & 16) ? 8 : 0;
  const int ln = lane & 15;
#pragma unroll
  for (int nt = 0; nt < 4; ++nt) {
#pragma unroll
    for (int r = 0; r < 8; ++r) {
      const int row = m0 + wave * 16 + r + mh;
      const int bb  = row >> 11;
      const int ss  = row & 2047;
      const int col = nt * 16 + ln;
      Y[(((size_t)bb * kH + h) * kS + ss) * kDK + col] = (_Float16)acc[nt][r];
    }
  }
}

// ---------------------------------------------------------------------------
// Kernel 2: flash attention. grid = (kS/64, kB*kH), block = 128 (4 waves).
// Q/K/V chunks staged with TENSOR_LOAD_TO_LDS (TDM), padded into kLdQ rows.
// ---------------------------------------------------------------------------
__global__ void __launch_bounds__(128)
attn_kernel(const _Float16* __restrict__ qh, const _Float16* __restrict__ kh,
            const _Float16* __restrict__ vh, const float* __restrict__ mask,
            _Float16* __restrict__ monolith) {
  __shared__ alignas(16) _Float16 ldsQ[64 * kLdQ];
  __shared__ alignas(16) _Float16 ldsK[64 * kLdQ];
  __shared__ alignas(16) _Float16 ldsV[64 * kLdQ];
  __shared__ alignas(16) _Float16 ldsP[64 * kLdQ];
  __shared__ float ldsS[64 * kLdS];
  __shared__ float m_s[64], l_s[64], rs_s[64];

  const int t    = threadIdx.x;
  const int wave = t >> 5, lane = t & 31;
  const int ln   = lane & 15;
  const int mh   = (lane & 16) ? 8 : 0;

  const int q0 = blockIdx.x * 64;
  const int bh = blockIdx.y;
  const int bb = bh >> 4, h = bh & 15;

  const _Float16* Qb = qh + (size_t)bh * kS * kDK;
  const _Float16* Kb = kh + (size_t)bh * kS * kDK;
  const _Float16* Vb = vh + (size_t)bh * kS * kDK;

  // Q block 64x64 f16 via TDM (wave 0 issues; barrier publishes)
  if (wave == 0) {
    tdm_load_2d(lds_off(ldsQ), Qb + (size_t)q0 * kDK, 64, 64, kDK, kS);
    wait_tensor();
  }
  if (t < 64) { m_s[t] = -1e30f; l_s[t] = 0.0f; }
  __syncthreads();

  // Q fragments are key-loop invariant — hoist.
  v16h qa[2];
#pragma unroll
  for (int ks = 0; ks < 2; ++ks)
    qa[ks] = frag_a(ldsQ, kLdQ, wave * 16, ks * 32, lane);

  v8f o[4] = {};
  const float scale = 0.125f;  // 1/sqrt(64)

  for (int j0 = 0; j0 < kS; j0 += 64) {
    // K & V chunks via TDM
    if (wave == 0) {
      tdm_load_2d(lds_off(ldsK), Kb + (size_t)j0 * kDK, 64, 64, kDK, kS);
      tdm_load_2d(lds_off(ldsV), Vb + (size_t)j0 * kDK, 64, 64, kDK, kS);
      wait_tensor();
    }
    if (j0 + 64 < kS) {
      __builtin_prefetch(Kb + (size_t)(j0 + 64) * kDK + t * 32, 0, 1);
      __builtin_prefetch(Vb + (size_t)(j0 + 64) * kDK + t * 32, 0, 1);
    }
    __syncthreads();

    // scores = Q * K^T (K chunk [key][dk] is n-major for B)
    v8f sc[4] = {};
#pragma unroll
    for (int nt = 0; nt < 4; ++nt) {
#pragma unroll
      for (int ks = 0; ks < 2; ++ks) {
        const v16h bk = frag_b_nmajor(ldsK, kLdQ, nt * 16, ks * 32, lane);
        sc[nt] = __builtin_amdgcn_wmma_f32_16x16x32_f16(
            false, qa[ks], false, bk, (short)0, sc[nt], false, false);
      }
    }
    // scale + mask -> score tile
#pragma unroll
    for (int nt = 0; nt < 4; ++nt) {
#pragma unroll
      for (int r = 0; r < 8; ++r) {
        const int row = wave * 16 + r + mh;
        const int col = nt * 16 + ln;
        ldsS[row * kLdS + col] = sc[nt][r] * scale
            + mask[(size_t)(q0 + row) * kS + j0 + col];
      }
    }
    __syncthreads();

    // online softmax: one thread per query row
    if (t < 64) {
      const float mo = m_s[t];
      float cmax = -1e30f;
#pragma unroll 8
      for (int c = 0; c < 64; ++c) cmax = fmaxf(cmax, ldsS[t * kLdS + c]);
      const float mn = fmaxf(mo, cmax);
      const float f  = __expf(mo - mn);
      float sum = 0.0f;
#pragma unroll 8
      for (int c = 0; c < 64; ++c) {
        const float p = __expf(ldsS[t * kLdS + c] - mn);
        ldsP[t * kLdQ + c] = (_Float16)p;
        sum += p;
      }
      m_s[t]  = mn;
      l_s[t]  = l_s[t] * f + sum;
      rs_s[t] = f;
    }
    __syncthreads();

    // rescale O, then O += P * V
    float fr[8];
#pragma unroll
    for (int r = 0; r < 8; ++r) fr[r] = rs_s[wave * 16 + r + mh];
#pragma unroll
    for (int nt = 0; nt < 4; ++nt)
#pragma unroll
      for (int r = 0; r < 8; ++r) o[nt][r] *= fr[r];

#pragma unroll
    for (int ks = 0; ks < 2; ++ks) {
      const v16h pa = frag_a(ldsP, kLdQ, wave * 16, ks * 32, lane);
#pragma unroll
      for (int nt = 0; nt < 4; ++nt) {
        const v16h bv = frag_b_kmajor(ldsV, kLdQ, nt * 16, ks * 32, lane);
        o[nt] = __builtin_amdgcn_wmma_f32_16x16x32_f16(
            false, pa, false, bv, (short)0, o[nt], false, false);
      }
    }
    __syncthreads();
  }

  // normalize + store to monolith [B, S, H*DK] f16
  float inv[8];
#pragma unroll
  for (int r = 0; r < 8; ++r) inv[r] = 1.0f / l_s[wave * 16 + r + mh];
#pragma unroll
  for (int nt = 0; nt < 4; ++nt)
#pragma unroll
    for (int r = 0; r < 8; ++r) {
      const int row = wave * 16 + r + mh;
      const int col = nt * 16 + ln;
      monolith[((size_t)bb * kS + q0 + row) * kD + h * kDK + col] =
          (_Float16)(o[nt][r] * inv[r]);
    }
}

// ---------------------------------------------------------------------------
// Kernel 3: out = monolith(f16) @ WO(f32->f16), fp32 result.
// A tile staged with async DMA; WO staged via float4 clauses.
// grid = (kM/64, kD/64), block = 128.
// ---------------------------------------------------------------------------
__global__ void __launch_bounds__(128)
out_gemm_kernel(const _Float16* __restrict__ monolith,
                const float* __restrict__ WO, float* __restrict__ out) {
  __shared__ alignas(16) _Float16 ldsA[64 * kLdQ];
  __shared__ alignas(16) _Float16 ldsBt[64 * kLdQ];

  const int t    = threadIdx.x;
  const int wave = t >> 5, lane = t & 31;
  const int m0   = blockIdx.x * 64;
  const int n0   = blockIdx.y * 64;

  v8f acc[4] = {};

  for (int kk = 0; kk < kD; kk += 64) {
    // A tile 64x64 f16 via async DMA (16B pieces)
#pragma unroll
    for (int i = 0; i < 4; ++i) {
      const int idx = t + i * 128;           // 0..511
      const int r = idx >> 3, c8 = idx & 7;
      const char* src =
          (const char*)(monolith + (size_t)(m0 + r) * kD + kk) + c8 * 16;
      async_copy_b128(lds_off(&ldsA[r * kLdQ + c8 * 8]), src);
    }
    // WO tile 64(d) x 64(n) fp32: batched float4 loads, transposed stores
    const float4* Wv = (const float4*)(WO + (size_t)kk * kD + n0);
    float4 tb[8];
#pragma unroll
    for (int i = 0; i < 8; ++i) {
      const int idx = t + i * 128;
      const int d = idx >> 4, c4 = idx & 15;
      tb[i] = Wv[(size_t)d * (kD / 4) + c4];
    }
#pragma unroll
    for (int i = 0; i < 8; ++i) {
      const int idx = t + i * 128;
      const int d = idx >> 4, c4 = idx & 15;
      ldsBt[(c4 * 4 + 0) * kLdQ + d] = (_Float16)tb[i].x;
      ldsBt[(c4 * 4 + 1) * kLdQ + d] = (_Float16)tb[i].y;
      ldsBt[(c4 * 4 + 2) * kLdQ + d] = (_Float16)tb[i].z;
      ldsBt[(c4 * 4 + 3) * kLdQ + d] = (_Float16)tb[i].w;
    }
    wait_async();
    __syncthreads();

#pragma unroll
    for (int ks = 0; ks < 2; ++ks) {
      const v16h a = frag_a(ldsA, kLdQ, wave * 16, ks * 32, lane);
#pragma unroll
      for (int nt = 0; nt < 4; ++nt) {
        const v16h bfr = frag_b_nmajor(ldsBt, kLdQ, nt * 16, ks * 32, lane);
        acc[nt] = __builtin_amdgcn_wmma_f32_16x16x32_f16(
            false, a, false, bfr, (short)0, acc[nt], false, false);
      }
    }
    __syncthreads();
  }

  const int mh = (lane & 16) ? 8 : 0;
  const int ln = lane & 15;
#pragma unroll
  for (int nt = 0; nt < 4; ++nt)
#pragma unroll
    for (int r = 0; r < 8; ++r) {
      const int row = m0 + wave * 16 + r + mh;
      out[(size_t)row * kD + n0 + nt * 16 + ln] = acc[nt][r];
    }
}

// ---------------------------------------------------------------------------
extern "C" void kernel_launch(void* const* d_in, const int* in_sizes, int n_in,
                              void* d_out, int out_size, void* d_ws, size_t ws_size,
                              hipStream_t stream) {
  const float* Q    = (const float*)d_in[0];
  const float* K    = (const float*)d_in[1];
  const float* V    = (const float*)d_in[2];
  const float* mask = (const float*)d_in[3];
  const float* WQ   = (const float*)d_in[4];
  const float* WK   = (const float*)d_in[5];
  const float* WV   = (const float*)d_in[6];
  const float* WO   = (const float*)d_in[7];
  float* out = (float*)d_out;

  const size_t per = (size_t)kB * kH * kS * kDK;   // 8M f16 elems = 16 MB
  _Float16* qh   = (_Float16*)d_ws;
  _Float16* kh   = qh + per;
  _Float16* vh   = kh + per;
  _Float16* mono = vh + per;                       // [B,S,kD] f16

  proj_kernel<<<dim3(kM / 64, kH, 3), dim3(128), 0, stream>>>(
      Q, K, V, WQ, WK, WV, qh, kh, vh);
  attn_kernel<<<dim3(kS / 64, kB * kH), dim3(128), 0, stream>>>(
      qh, kh, vh, mask, mono);
  out_gemm_kernel<<<dim3(kM / 64, kD / 64), dim3(128), 0, stream>>>(
      mono, WO, out);
}
